// IMultiHeadAttention2D_4415226380755
// MI455X (gfx1250) — matrix operand order repaired and checked by reading, not compiled, not run on hardware
//
#include <hip/hip_runtime.h>
#include <hip/hip_bf16.h>
#include <math.h>

// ---------------------------------------------------------------------------
// MI455X (gfx1250) fused multi-head attention, wave32 + v_wmma_f32_16x16x32_f16
//
// B=8, C=256, n=1024, heads=8, head_dim=64, hidden=512.  ~26 GFLOP vs ~16MB
// true HBM traffic -> compute bound -> all GEMMs through K=32 f16 WMMA (f32
// accum), flash-style fused softmax (no 256MB logits tensor).
//
// Data movement: K/V/attn-out written once in WMMA-B-fragment-native layout
// ([frag][lane][16 halfs]) and consumed as b128 loads.  Shared panels are
// staged into LDS with GLOBAL_LOAD_ASYNC_TO_LDS_B128 (ASYNCcnt), double
// buffered in the attention kernel so the DMA overlaps the 16 WMMAs/step.
// ---------------------------------------------------------------------------

typedef _Float16 half_t;
typedef __attribute__((ext_vector_type(16))) _Float16 v16h;
typedef __attribute__((ext_vector_type(8)))  _Float16 v8h;
typedef __attribute__((ext_vector_type(8)))  float    v8f;
typedef __attribute__((ext_vector_type(4)))  int      v4i;

#define BB      8
#define CC      256
#define NN      1024
#define NH      8
#define HD      64
#define HIDDEN  512
#define ATTSCALE 0.125f

// ---- async global->LDS (gfx1250); falls back to sync copy if unavailable
#if defined(__gfx1250__) && __has_builtin(__builtin_amdgcn_global_load_async_to_lds_b128)
#define ASYNC_COPY16(src, dst)                                            \
    __builtin_amdgcn_global_load_async_to_lds_b128(                        \
        (__attribute__((address_space(1))) v4i*)(src),                     \
        (__attribute__((address_space(3))) v4i*)(dst), 0, 0)
#define ASYNC_WAIT() asm volatile("s_wait_asynccnt 0" ::: "memory")
#else
#define ASYNC_COPY16(src, dst) (*(v8h*)(dst) = *(const v8h*)(src))
#define ASYNC_WAIT()
#endif

static __device__ __forceinline__ v8f wmma32(v16h a, v16h b, v8f c) {
    return __builtin_amdgcn_wmma_f32_16x16x32_f16(false, a, false, b,
                                                  (short)0, c, false, false);
}
static __device__ __forceinline__ v16h cat8(v8h lo, v8h hi) {
    v16h r;
    #pragma unroll
    for (int i = 0; i < 8; ++i) { r[i] = lo[i]; r[i + 8] = hi[i]; }
    return r;
}

// ---------------------------------------------------------------------------
// Kernel 1: qkv[o,n] = sum_c Wqkv[o,c]*x[b,c,n]
// Block = 8 waves sharing one 16-col n-tile (x panel staged once into LDS in
// B-fragment layout; f32->f16 conversion forces the manual staging path).
// Outputs: Qt[bh,n,d] row-major; K,V fragment-native.
// ---------------------------------------------------------------------------
__global__ __launch_bounds__(256) void qkv_proj(const float* __restrict__ x,
                                                const float* __restrict__ wqkv,
                                                half_t* __restrict__ qt,
                                                half_t* __restrict__ kmf,
                                                half_t* __restrict__ vtf) {
    __shared__ half_t xs[8 * 512];
    const int tid  = threadIdx.x;
    const int lane = tid & 31;
    const int wave = tid >> 5;
    const int l16  = lane & 15;
    const int hs   = lane >> 4;

    const int b   = blockIdx.x / (12 * 64);
    const int rem = blockIdx.x % (12 * 64);
    const int og  = rem / 64;
    const int nt  = rem % 64;
    const int n0  = nt * 16;

    const float* xb = x + (size_t)b * CC * NN;

    {   // stage x panel (256K x 16n) into LDS fragment layout
        const int k = tid;
        const float4* src = (const float4*)(xb + (size_t)k * NN + n0);
        float4 f0 = src[0], f1 = src[1], f2 = src[2], f3 = src[3];
        float vals[16] = { f0.x,f0.y,f0.z,f0.w, f1.x,f1.y,f1.z,f1.w,
                           f2.x,f2.y,f2.z,f2.w, f3.x,f3.y,f3.z,f3.w };
        const int kg = k >> 5, e = (k & 7) | (((k >> 4) & 1) << 3), lh = (k >> 3) & 1;
        half_t* dst = xs + kg * 512 + e;
        #pragma unroll
        for (int nl = 0; nl < 16; ++nl)
            dst[(nl + 16 * lh) * 16] = (half_t)vals[nl];
    }
    __syncthreads();

    const int o0 = og * 128 + wave * 16;
    const float* arow = wqkv + (size_t)(o0 + l16) * CC;

    v8f acc = {};
    #pragma unroll 1
    for (int ks = 0; ks < 8; ++ks) {
        const int k0 = ks * 32;
        v16h af;
        {
            const float4 w0 = *(const float4*)(arow + k0 + 8 * hs);
            const float4 w1 = *(const float4*)(arow + k0 + 8 * hs + 4);
            const float4 w2 = *(const float4*)(arow + k0 + 8 * hs + 16);
            const float4 w3 = *(const float4*)(arow + k0 + 8 * hs + 20);
            af[0]=(half_t)w0.x; af[1]=(half_t)w0.y; af[2]=(half_t)w0.z; af[3]=(half_t)w0.w;
            af[4]=(half_t)w1.x; af[5]=(half_t)w1.y; af[6]=(half_t)w1.z; af[7]=(half_t)w1.w;
            af[8]=(half_t)w2.x; af[9]=(half_t)w2.y; af[10]=(half_t)w2.z; af[11]=(half_t)w2.w;
            af[12]=(half_t)w3.x; af[13]=(half_t)w3.y; af[14]=(half_t)w3.z; af[15]=(half_t)w3.w;
        }
        const v16h bf = *(const v16h*)(xs + ks * 512 + lane * 16);
        acc = wmma32(af, bf, acc);
    }

    const int s     = o0 >> 9;
    const int hh    = (o0 >> 6) & 7;
    const int dbase = o0 & 63;
    const int n     = n0 + l16;
    const int bh    = b * NH + hh;
    v8h pk;
    #pragma unroll
    for (int r = 0; r < 8; ++r) pk[r] = (half_t)acc[r];

    if (s == 0) {
        *(v8h*)(qt + ((size_t)bh * NN + n) * HD + dbase + 8 * hs) = pk;
    } else if (s == 1) {
        const int dg = dbase >> 5;
        const int lp = l16 + 16 * hs;
        *(v8h*)(kmf + (((size_t)bh * 64 + nt) * 2 + dg) * 512 + lp * 16 + 8 * ((dbase >> 4) & 1)) = pk;
    } else {
        const int t  = (dbase >> 4) & 3;
        const int jg = n0 >> 5;
        const int jh = (n0 >> 4) & 1;
        half_t* base = vtf + (((size_t)bh * 32 + jg) * 4 + t) * 512;
        #pragma unroll
        for (int r = 0; r < 8; ++r)
            base[((r + 8 * hs) + 16 * ((l16 >> 3) & 1)) * 16 + (l16 & 7) + 8 * jh] = pk[r];
    }
}

// ---------------------------------------------------------------------------
// Kernel 2: fused attention.  Block = 8 waves (8 query row-tiles) sharing one
// head-batch j-stream.  Each 64-key step: K/V panels (16KB) arrive in LDS via
// double-buffered async DMA; 16 WMMAs consume them while the next panel is in
// flight.  One barrier per step.
// ---------------------------------------------------------------------------
__global__ __launch_bounds__(256) void attn(const half_t* __restrict__ qt,
                                            const half_t* __restrict__ kmf,
                                            const half_t* __restrict__ vtf,
                                            half_t* __restrict__ aof) {
    __shared__ half_t kvbuf[2][8192];       // [buf][K frags 0-7 | V frags 8-15]
    __shared__ half_t pbuf[8][1024];        // per-wave 16x64 P tile
    const int tid  = threadIdx.x;
    const int lane = tid & 31;
    const int wave = tid >> 5;
    const int l16  = lane & 15;
    const int hs   = lane >> 4;
    const int bh   = blockIdx.x >> 3;
    const int i0   = ((blockIdx.x & 7) * 8 + wave) * 16;
    const int b    = bh >> 3, h = bh & 7;

    const half_t* qrow  = qt + ((size_t)bh * NN + i0 + l16) * HD;
    const half_t* kbase = kmf + (size_t)bh * 64 * 1024;   // 64 nt * 2 dg * 512
    const half_t* vbase = vtf + (size_t)bh * 32 * 2048;   // 32 jg * 4 t * 512
    half_t* pw = pbuf[wave];

    const v16h a0 = cat8(*(const v8h*)(qrow + 8 * hs),
                         *(const v8h*)(qrow + 8 * hs + 16));
    const v16h a1 = cat8(*(const v8h*)(qrow + 32 + 8 * hs),
                         *(const v8h*)(qrow + 32 + 8 * hs + 16));

    float m[8], lsum[8];
    #pragma unroll
    for (int r = 0; r < 8; ++r) { m[r] = -1e30f; lsum[r] = 0.0f; }
    v8f oacc[4] = {};

    // prologue: stage panel 0
    {
        const half_t* ksrc = kbase + (size_t)tid * 16;
        const half_t* vsrc = vbase + (size_t)tid * 16;
        half_t* kdst = &kvbuf[0][tid * 16];
        half_t* vdst = &kvbuf[0][4096 + tid * 16];
        ASYNC_COPY16(ksrc, kdst);      ASYNC_COPY16(ksrc + 8, kdst + 8);
        ASYNC_COPY16(vsrc, vdst);      ASYNC_COPY16(vsrc + 8, vdst + 8);
        ASYNC_WAIT();
    }
    __syncthreads();

    #pragma unroll 1
    for (int jt = 0; jt < 16; ++jt) {       // 64 keys per step
        const int cur = jt & 1;
        if (jt + 1 < 16) {                   // kick off next panel (no wait)
            const half_t* ksrc = kbase + (size_t)(jt + 1) * 4096 + tid * 16;
            const half_t* vsrc = vbase + (size_t)(jt + 1) * 4096 + tid * 16;
            half_t* kdst = &kvbuf[cur ^ 1][tid * 16];
            half_t* vdst = &kvbuf[cur ^ 1][4096 + tid * 16];
            ASYNC_COPY16(ksrc, kdst);  ASYNC_COPY16(ksrc + 8, kdst + 8);
            ASYNC_COPY16(vsrc, vdst);  ASYNC_COPY16(vsrc + 8, vdst + 8);
        }
        const half_t* kl = &kvbuf[cur][0];

        // S = Qt(16x64) x K(64x64): 4 col-tiles, 8 WMMAs
        v8f s[4];
        #pragma unroll
        for (int c = 0; c < 4; ++c) {
            const v16h k0f = *(const v16h*)(kl + (2 * c) * 512 + lane * 16);
            const v16h k1f = *(const v16h*)(kl + (2 * c + 1) * 512 + lane * 16);
            v8f z = {};
            s[c] = wmma32(a0, k0f, z);
            s[c] = wmma32(a1, k1f, s[c]);
        }

        // online softmax over 64 cols (row stats replicated per 16-lane half)
        float p[4][8], alpha[8];
        #pragma unroll
        for (int r = 0; r < 8; ++r) {
            float x0 = s[0][r] * ATTSCALE, x1 = s[1][r] * ATTSCALE;
            float x2 = s[2][r] * ATTSCALE, x3 = s[3][r] * ATTSCALE;
            float cmax = fmaxf(fmaxf(x0, x1), fmaxf(x2, x3));
            cmax = fmaxf(cmax, __shfl_xor(cmax, 1));
            cmax = fmaxf(cmax, __shfl_xor(cmax, 2));
            cmax = fmaxf(cmax, __shfl_xor(cmax, 4));
            cmax = fmaxf(cmax, __shfl_xor(cmax, 8));
            const float mn = fmaxf(m[r], cmax);
            alpha[r] = __expf(m[r] - mn);
            m[r] = mn;
            p[0][r] = __expf(x0 - mn); p[1][r] = __expf(x1 - mn);
            p[2][r] = __expf(x2 - mn); p[3][r] = __expf(x3 - mn);
            float rs = (p[0][r] + p[1][r]) + (p[2][r] + p[3][r]);
            rs += __shfl_xor(rs, 1);
            rs += __shfl_xor(rs, 2);
            rs += __shfl_xor(rs, 4);
            rs += __shfl_xor(rs, 8);
            lsum[r] = lsum[r] * alpha[r] + rs;
        }
        #pragma unroll
        for (int t = 0; t < 4; ++t)
            #pragma unroll
            for (int r = 0; r < 8; ++r)
                oacc[t][r] *= alpha[r];

        // P tile -> per-wave LDS region (same-wave DS ordering, no barrier)
        #pragma unroll
        for (int c = 0; c < 4; ++c)
            #pragma unroll
            for (int r = 0; r < 8; ++r)
                pw[(r + 8 * hs) * 64 + c * 16 + l16] = (half_t)p[c][r];
        const v16h pa0 = cat8(*(const v8h*)(pw + l16 * 64 + 8 * hs),
                              *(const v8h*)(pw + l16 * 64 + 16 + 8 * hs));
        const v16h pa1 = cat8(*(const v8h*)(pw + l16 * 64 + 32 + 8 * hs),
                              *(const v8h*)(pw + l16 * 64 + 48 + 8 * hs));

        // O += P(16x64) x Vt(64x64): 8 WMMAs
        #pragma unroll
        for (int t = 0; t < 4; ++t) {
            const v16h vb0 = *(const v16h*)(kl + (8 + t) * 512 + lane * 16);
            const v16h vb1 = *(const v16h*)(kl + (12 + t) * 512 + lane * 16);
            oacc[t] = wmma32(pa0, vb0, oacc[t]);
            oacc[t] = wmma32(pa1, vb1, oacc[t]);
        }

        ASYNC_WAIT();          // own next-panel DMA done
        __syncthreads();       // all waves done with current panel
    }

    // normalize + store attn-out in fragment-native layout
    float linv[8];
    #pragma unroll
    for (int r = 0; r < 8; ++r) linv[r] = 1.0f / lsum[r];
    const int ntile = i0 >> 4;
    #pragma unroll
    for (int t = 0; t < 4; ++t) {
        const int kg = 2 * h + (t >> 1);
        half_t* base = aof + (((size_t)b * 64 + ntile) * 16 + kg) * 512;
        #pragma unroll
        for (int r = 0; r < 8; ++r)
            base[((r + 8 * hs) + 16 * ((l16 >> 3) & 1)) * 16 + (l16 & 7) + 8 * (t & 1)] =
                (half_t)(oacc[t][r] * linv[r]);
    }
}

// ---------------------------------------------------------------------------
// Kernel 3: out[b,c,n] = sum_o Wout[c,o]*ao[b,o,n] + bout[c]
// Block = 8 waves sharing one n-tile; the whole 16KB B panel is async-staged
// into LDS once and consumed by all waves.
// ---------------------------------------------------------------------------
__global__ __launch_bounds__(256) void out_proj(const half_t* __restrict__ aof,
                                                const float* __restrict__ wout,
                                                const float* __restrict__ bout,
                                                float* __restrict__ out) {
    __shared__ half_t bs[16 * 512];         // 16 k-steps of B fragments
    const int tid  = threadIdx.x;
    const int lane = tid & 31;
    const int wave = tid >> 5;
    const int l16  = lane & 15;
    const int hs   = lane >> 4;

    const int b   = blockIdx.x >> 7;        // B * 64 nt * 2 cg = 1024 blocks
    const int rem = blockIdx.x & 127;
    const int nt  = rem >> 1;
    const int cg  = rem & 1;
    const int n0  = nt * 16;
    const int c0  = cg * 128 + wave * 16;

    {   // async-stage contiguous 16KB fragment panel
        const half_t* src = aof + ((size_t)b * 64 + nt) * 16 * 512 + tid * 32;
        half_t* dst = bs + tid * 32;
        ASYNC_COPY16(src,      dst);
        ASYNC_COPY16(src + 8,  dst + 8);
        ASYNC_COPY16(src + 16, dst + 16);
        ASYNC_COPY16(src + 24, dst + 24);
        ASYNC_WAIT();
    }
    __syncthreads();

    const float* arow = wout + (size_t)(c0 + l16) * HIDDEN;

    v8f acc = {};
    #pragma unroll 1
    for (int ks = 0; ks < 16; ++ks) {
        const int k0 = ks * 32;
        if (ks + 1 < 16) __builtin_prefetch(arow + k0 + 32, 0, 3);
        v16h af;
        {
            const float4 w0 = *(const float4*)(arow + k0 + 8 * hs);
            const float4 w1 = *(const float4*)(arow + k0 + 8 * hs + 4);
            const float4 w2 = *(const float4*)(arow + k0 + 8 * hs + 16);
            const float4 w3 = *(const float4*)(arow + k0 + 8 * hs + 20);
            af[0]=(half_t)w0.x; af[1]=(half_t)w0.y; af[2]=(half_t)w0.z; af[3]=(half_t)w0.w;
            af[4]=(half_t)w1.x; af[5]=(half_t)w1.y; af[6]=(half_t)w1.z; af[7]=(half_t)w1.w;
            af[8]=(half_t)w2.x; af[9]=(half_t)w2.y; af[10]=(half_t)w2.z; af[11]=(half_t)w2.w;
            af[12]=(half_t)w3.x; af[13]=(half_t)w3.y; af[14]=(half_t)w3.z; af[15]=(half_t)w3.w;
        }
        const v16h bf = *(const v16h*)(bs + ks * 512 + lane * 16);
        acc = wmma32(af, bf, acc);
    }
    #pragma unroll
    for (int r = 0; r < 8; ++r) {
        const int c = c0 + r + 8 * hs;
        out[((size_t)(b * CC + c)) * NN + n0 + l16] = acc[r] + bout[c];
    }
}

// ---------------------------------------------------------------------------
extern "C" void kernel_launch(void* const* d_in, const int* in_sizes, int n_in,
                              void* d_out, int out_size, void* d_ws, size_t ws_size,
                              hipStream_t stream) {
    const float* x    = (const float*)d_in[0];
    const float* wqkv = (const float*)d_in[1];
    const float* wout = (const float*)d_in[2];
    const float* bout = (const float*)d_in[3];
    float* out = (float*)d_out;

    const size_t per = (size_t)BB * NH * NN * HD;   // 4M halfs each
    half_t* qt  = (half_t*)d_ws;
    half_t* kmf = qt  + per;
    half_t* vtf = kmf + per;
    half_t* aof = vtf + per;

    qkv_proj<<<6144, 256, 0, stream>>>(x, wqkv, qt, kmf, vtf);
    attn<<<512, 256, 0, stream>>>(qt, kmf, vtf, aof);
    out_proj<<<1024, 256, 0, stream>>>(aof, wout, bout, out);
}
